// Heterocoder_9191230013906
// MI455X (gfx1250) — matrix operand order repaired and checked by reading, not compiled, not run on hardware
//
#include <hip/hip_runtime.h>

// ---------------------------------------------------------------------------
// Types for WMMA fragments (gfx1250 wave32)
// ---------------------------------------------------------------------------
typedef __attribute__((ext_vector_type(16))) __bf16       bf16x16;
typedef __attribute__((ext_vector_type(8)))  float        f32x8;
typedef __attribute__((ext_vector_type(4)))  unsigned int u32x4;
typedef __attribute__((ext_vector_type(2)))  unsigned int u32x2;

union FragU {
    u32x4   q[2];
    bf16x16 v;
};

__device__ __forceinline__ unsigned short f2bf(float f) {
    // round-to-nearest-even fp32 -> bf16
    unsigned int u = __float_as_uint(f);
    unsigned int r = u + 0x7FFFu + ((u >> 16) & 1u);
    return (unsigned short)(r >> 16);
}

// A-matrix fragment (16x32 bf16) from LDS, per ISA 7.12.2:
// lanes 0-15: v0-3 -> K base+0..7, v4-7 -> K base+16..23
// lanes16-31: v0-3 -> K base+8..15, v4-7 -> K base+24..31
__device__ __forceinline__ bf16x16 a_frag(const unsigned short* s, int stride,
                                          int row, int kbase, int lane) {
    int kk = kbase + ((lane & 16) ? 8 : 0);
    const unsigned short* p = s + row * stride + kk;
    FragU f;
    f.q[0] = *(const u32x4*)(p);
    f.q[1] = *(const u32x4*)(p + 16);
    return f.v;
}

// B-matrix fragment (32x16 bf16) from bf16 weights stored transposed [N][K]:
// lane l holds column n = l%16; lanes 0-15 K=base+0..7,16..23, lanes 16-31 +8.
__device__ __forceinline__ bf16x16 b_frag(const unsigned short* wt, int K,
                                          int ncol, int kbase, int lane) {
    int k0 = kbase + ((lane & 16) ? 8 : 0);
    const unsigned short* p = wt + (size_t)ncol * K + k0;
    FragU f;
    f.q[0] = *(const u32x4*)(p);
    f.q[1] = *(const u32x4*)(p + 16);
    return f.v;
}

__device__ __forceinline__ f32x8 wmma_bf16(bf16x16 a, bf16x16 b, f32x8 c) {
    return __builtin_amdgcn_wmma_f32_16x16x32_bf16(false, a, false, b,
                                                   (short)0, c, false, false);
}

// ---------------------------------------------------------------------------
// Generic fused  (gather/concat) -> Linear(SiLU) -> Linear -> LayerNorm
//                -> residual store (+ optional scatter-add)
// MODE 0: edge MLP   (chunks: sender_x[src], receiver_x[dst], edge_attr)
// MODE 1: node MLP   (chunks: receiver_x, agg/cnt)
// MODE 2: sender MLP (chunk : sender_x)
// Block = 128 threads = 4 waves, M-tile = 64 rows, wave w owns rows 16w..16w+15
// ---------------------------------------------------------------------------
template <int MODE, int NCHUNK>
__global__ __launch_bounds__(128) void mlp_ln_kernel(
    const float* __restrict__ in0, const float* __restrict__ in1,
    const float* __restrict__ in2, const int* __restrict__ eidx,
    const float* __restrict__ cnt, const unsigned short* __restrict__ w1t,
    const float* __restrict__ b1, const unsigned short* __restrict__ w2t,
    const float* __restrict__ b2, const float* __restrict__ gamma,
    const float* __restrict__ beta, const float* __restrict__ resid,
    float* __restrict__ out, float* __restrict__ agg, int E) {
    constexpr int K1 = NCHUNK * 128;

    __shared__ __align__(16) unsigned short sIn[64][136];  // bf16 chunk stage
    __shared__ __align__(16) unsigned short sH[64][264];   // bf16 hidden (H=256)

    const int tid  = threadIdx.x;
    const int lane = tid & 31;
    const int wave = tid >> 5;
    const int m0   = blockIdx.x * 64;

    // ---- Phase 1: in @ w1  (K = NCHUNK*128, N = 256 -> 16 N-tiles) ----
    f32x8 acc[16] = {};

    for (int c = 0; c < NCHUNK; ++c) {
        // cooperative stage of one 64x128 f32 chunk -> bf16 LDS
        for (int i = tid; i < 64 * 32; i += 128) {
            int r  = i >> 5;
            int c4 = i & 31;
            const float* base;
            float scale = 1.0f;
            if (MODE == 0) {
                if (c == 0)      base = in0 + (size_t)eidx[m0 + r] * 128;
                else if (c == 1) base = in1 + (size_t)eidx[E + m0 + r] * 128;
                else             base = in2 + (size_t)(m0 + r) * 128;
            } else if (MODE == 1) {
                if (c == 0) base = in0 + (size_t)(m0 + r) * 128;
                else {
                    base  = in1 + (size_t)(m0 + r) * 128;
                    scale = 1.0f / fmaxf(cnt[m0 + r], 1.0f);  // mean aggregation
                }
            } else {
                base = in0 + (size_t)(m0 + r) * 128;
            }
            float4 v = ((const float4*)base)[c4];
            u32x2 t;
            t.x = (unsigned int)f2bf(v.x * scale) |
                  ((unsigned int)f2bf(v.y * scale) << 16);
            t.y = (unsigned int)f2bf(v.z * scale) |
                  ((unsigned int)f2bf(v.w * scale) << 16);
            *(u32x2*)&sIn[r][c4 * 4] = t;
        }
        __syncthreads();

#pragma unroll
        for (int ks = 0; ks < 4; ++ks) {
            bf16x16 a =
                a_frag(&sIn[0][0], 136, wave * 16 + (lane & 15), ks * 32, lane);
#pragma unroll
            for (int nt = 0; nt < 16; ++nt) {
                bf16x16 b = b_frag(w1t, K1, nt * 16 + (lane & 15),
                                   c * 128 + ks * 32, lane);
                acc[nt] = wmma_bf16(a, b, acc[nt]);
            }
        }
        __syncthreads();  // before next chunk overwrites sIn
    }

    // bias + SiLU -> bf16 hidden in LDS (C-layout: lane holds col lane%16,
    // rows v + 8*(lane>=16))
    {
        int mrow = wave * 16 + ((lane & 16) ? 8 : 0);
        int ncol = lane & 15;
#pragma unroll
        for (int nt = 0; nt < 16; ++nt) {
            int   n  = nt * 16 + ncol;
            float bb = b1[n];
#pragma unroll
            for (int v = 0; v < 8; ++v) {
                float x = acc[nt][v] + bb;
                float s = x / (1.0f + __expf(-x));  // SiLU
                sH[mrow + v][n] = f2bf(s);
            }
        }
    }
    // no barrier: each wave only reads back its own 16 rows of sH

    // ---- Phase 2: h @ w2  (K = 256, N = 128 -> 8 N-tiles) ----
    f32x8 acc2[8] = {};
#pragma unroll
    for (int ks = 0; ks < 8; ++ks) {
        bf16x16 a =
            a_frag(&sH[0][0], 264, wave * 16 + (lane & 15), ks * 32, lane);
#pragma unroll
        for (int nt = 0; nt < 8; ++nt) {
            bf16x16 b =
                b_frag(w2t, 256, nt * 16 + (lane & 15), ks * 32, lane);
            acc2[nt] = wmma_bf16(a, b, acc2[nt]);
        }
    }

    // ---- Phase 3: bias + LayerNorm (in registers) + residual / scatter ----
    {
        int ncol = lane & 15;
#pragma unroll
        for (int nt = 0; nt < 8; ++nt) {
            float bb = b2[nt * 16 + ncol];
#pragma unroll
            for (int v = 0; v < 8; ++v) acc2[nt][v] += bb;
        }

#pragma unroll
        for (int v = 0; v < 8; ++v) {
            float sum = 0.f, sq = 0.f;
#pragma unroll
            for (int nt = 0; nt < 8; ++nt) {
                float y = acc2[nt][v];
                sum += y;
                sq += y * y;
            }
            // reduce across the 16 lanes holding this row (masks < 16 keep
            // the xor-shuffle inside each half-wave group)
#pragma unroll
            for (int mask = 1; mask < 16; mask <<= 1) {
                sum += __shfl_xor(sum, mask, 32);
                sq  += __shfl_xor(sq,  mask, 32);
            }
            float mu   = sum * (1.0f / 128.0f);
            float var  = sq * (1.0f / 128.0f) - mu * mu;
            float rstd = rsqrtf(var + 1e-5f);

            int row  = m0 + wave * 16 + ((lane & 16) ? 8 : 0) + v;
            int dsti = 0;
            if (MODE == 0) dsti = eidx[E + row];

#pragma unroll
            for (int nt = 0; nt < 8; ++nt) {
                int   n = nt * 16 + ncol;
                float p = (acc2[nt][v] - mu) * rstd * gamma[n] + beta[n];
                out[(size_t)row * 128 + n] = resid[(size_t)row * 128 + n] + p;
                if (MODE == 0)
                    atomicAdd(&agg[(size_t)dsti * 128 + n], p);  // scatter-sum
            }
        }
    }
}

// ---------------------------------------------------------------------------
// Helper kernels
// ---------------------------------------------------------------------------
__global__ void zero_kernel(float* p, long n) {
    for (long i = (long)blockIdx.x * blockDim.x + threadIdx.x; i < n;
         i += (long)gridDim.x * blockDim.x)
        p[i] = 0.0f;
}

__global__ void count_kernel(const int* __restrict__ eidx,
                             float* __restrict__ cnt, int E) {
    int e = blockIdx.x * blockDim.x + threadIdx.x;
    if (e < E) atomicAdd(&cnt[eidx[E + e]], 1.0f);
}

// convert W [K][N] f32 (row-major) -> bf16 transposed [N][K]
__global__ void wt_kernel(const float* __restrict__ w,
                          unsigned short* __restrict__ wt, int K, int N) {
    int i = blockIdx.x * blockDim.x + threadIdx.x;
    if (i < K * N) {
        int k       = i / N;
        int n       = i - k * N;
        wt[(size_t)n * K + k] = f2bf(w[i]);
    }
}

// ---------------------------------------------------------------------------
// Launch
// ---------------------------------------------------------------------------
extern "C" void kernel_launch(void* const* d_in, const int* in_sizes, int n_in,
                              void* d_out, int out_size, void* d_ws,
                              size_t ws_size, hipStream_t stream) {
    constexpr int N = 40000, E = 640000;

    const float* sender_x   = (const float*)d_in[0];
    const float* receiver_x = (const float*)d_in[1];
    const float* edge_attr  = (const float*)d_in[2];
    const int*   eidx       = (const int*)d_in[3];
    const float* ew1 = (const float*)d_in[4];
    const float* eb1 = (const float*)d_in[5];
    const float* ew2 = (const float*)d_in[6];
    const float* eb2 = (const float*)d_in[7];
    const float* eg  = (const float*)d_in[8];
    const float* ebt = (const float*)d_in[9];
    const float* nw1 = (const float*)d_in[10];
    const float* nb1 = (const float*)d_in[11];
    const float* nw2 = (const float*)d_in[12];
    const float* nb2 = (const float*)d_in[13];
    const float* ng  = (const float*)d_in[14];
    const float* nbt = (const float*)d_in[15];
    const float* sw1 = (const float*)d_in[16];
    const float* sb1 = (const float*)d_in[17];
    const float* sw2 = (const float*)d_in[18];
    const float* sb2 = (const float*)d_in[19];
    const float* sg  = (const float*)d_in[20];
    const float* sbt = (const float*)d_in[21];

    float* sender_out   = (float*)d_out;
    float* receiver_out = sender_out + (size_t)N * 128;
    float* edge_out     = receiver_out + (size_t)N * 128;

    // workspace layout: 6 bf16 transposed weights, then agg[N*128], cnt[N]
    unsigned short* ew1t = (unsigned short*)d_ws;     // 384x256
    unsigned short* ew2t = ew1t + 98304;              // 256x128
    unsigned short* nw1t = ew2t + 32768;              // 256x256
    unsigned short* nw2t = nw1t + 65536;              // 256x128
    unsigned short* sw1t = nw2t + 32768;              // 128x256
    unsigned short* sw2t = sw1t + 32768;              // 256x128
    float*          agg  = (float*)(sw2t + 32768);    // byte offset 589824
    float*          cnt  = agg + (size_t)N * 128;

    // 1) zero scatter accumulators (agg + cnt)
    zero_kernel<<<2048, 256, 0, stream>>>(agg, (long)N * 128 + N);

    // 2) weight conversion/transpose to bf16 [out][in]
    wt_kernel<<<(98304 + 255) / 256, 256, 0, stream>>>(ew1, ew1t, 384, 256);
    wt_kernel<<<(32768 + 255) / 256, 256, 0, stream>>>(ew2, ew2t, 256, 128);
    wt_kernel<<<(65536 + 255) / 256, 256, 0, stream>>>(nw1, nw1t, 256, 256);
    wt_kernel<<<(32768 + 255) / 256, 256, 0, stream>>>(nw2, nw2t, 256, 128);
    wt_kernel<<<(32768 + 255) / 256, 256, 0, stream>>>(sw1, sw1t, 128, 256);
    wt_kernel<<<(32768 + 255) / 256, 256, 0, stream>>>(sw2, sw2t, 256, 128);

    // 3) receiver in-degree counts
    count_kernel<<<(E + 255) / 256, 256, 0, stream>>>(eidx, cnt, E);

    // 4) edge MLP (+ edge residual out, + scatter-add of messages into agg)
    mlp_ln_kernel<0, 3><<<E / 64, 128, 0, stream>>>(
        sender_x, receiver_x, edge_attr, eidx, nullptr, ew1t, eb1, ew2t, eb2,
        eg, ebt, edge_attr, edge_out, agg, E);

    // 5) node MLP on [receiver_x | agg/cnt] (+ receiver residual out)
    mlp_ln_kernel<1, 2><<<N / 64, 128, 0, stream>>>(
        receiver_x, agg, nullptr, nullptr, cnt, nw1t, nb1, nw2t, nb2, ng, nbt,
        receiver_x, receiver_out, nullptr, 0);

    // 6) sender MLP (+ sender residual out)
    mlp_ln_kernel<2, 1><<<N / 64, 128, 0, stream>>>(
        sender_x, nullptr, nullptr, nullptr, nullptr, sw1t, sb1, sw2t, sb2, sg,
        sbt, sender_x, sender_out, nullptr, 0);
}
